// SpikingSelfAttention_78752520339710
// MI455X (gfx1250) — 16-bit and 8-bit matrix operand order repaired and checked by reading, compiled, not run on hardware
//
#include <hip/hip_runtime.h>
#include <hip/hip_bf16.h>

typedef __attribute__((ext_vector_type(16))) _Float16 v16h;
typedef __attribute__((ext_vector_type(8)))  float    v8f;
typedef __attribute__((ext_vector_type(8)))  int      v8i;
typedef __attribute__((ext_vector_type(4)))  unsigned u32x4;
typedef __attribute__((ext_vector_type(8)))  int      i32x8;
typedef __attribute__((ext_vector_type(4)))  int      i32x4;

#define BB   32
#define CC   512
#define T0   4096
#define TP   512
#define HH   8
#define DD   64
#define O3   1536

#if defined(__has_builtin)
#  if __has_builtin(__builtin_amdgcn_tensor_load_to_lds) && \
      __has_builtin(__builtin_amdgcn_s_wait_tensorcnt)
#    define HAVE_TDM 1
#  endif
#endif
#if defined(__has_include)
#  if __has_include(<hip/amd_detail/amd_gfx1250_TDM.h>)
#    define TDM_6ARG 1
#  endif
#endif

#if defined(HAVE_TDM)
#  if defined(TDM_6ARG)
#    warning "CDNA5 probe: TDM path ACTIVE (6-arg tensor_load_to_lds)"
#  else
#    warning "CDNA5 probe: TDM path ACTIVE (5-arg tensor_load_to_lds)"
#  endif
#else
#  warning "CDNA5 probe: TDM builtins NOT available - manual LDS staging fallback"
#endif


__device__ __forceinline__ v16h a_frag16(const _Float16* rowp, int k0h, int hi) {
  union { v16h v; unsigned u[8]; } f;
  const unsigned* r = (const unsigned*)rowp;
  int kd = k0h >> 1;
#pragma unroll
  for (int j = 0; j < 8; ++j) f.u[j] = r[kd + j + 8 * hi];
  return f.v;
}

__device__ __forceinline__ v16h b_frag16(const _Float16* rowp, int k0h, int hi) {
  union { v16h v; unsigned u[8]; } f;
  const unsigned* r = (const unsigned*)rowp;
  int kd = k0h >> 1;
#pragma unroll
  for (int v = 0; v < 8; ++v) f.u[v] = r[kd + v + 8 * hi];
  return f.v;
}

__device__ __forceinline__ v16h b_frag16_i8(const unsigned char* rowp, int k0, int hi) {
  union { v16h v; _Float16 h[16]; } f;
#pragma unroll
  for (int i = 0; i < 16; ++i) f.h[i] = (_Float16)(float)rowp[k0 + i + 16 * hi];
  return f.v;
}

__device__ __forceinline__ v8i a_frag8(const unsigned* rowdw, int s0dw, int hi) {
  v8i f;
#pragma unroll
  for (int j = 0; j < 8; ++j) {
    int kb = (j & 1) + ((j >> 1) & 1) * 4 + (j >> 2) * 8 + hi * 2;
    f[j] = (int)rowdw[s0dw + kb];
  }
  return f;
}

__device__ __forceinline__ v8i b_frag8(const unsigned* rowdw, int s0dw, int hi) {
  v8i f;
#pragma unroll
  for (int v = 0; v < 8; ++v) {
    int kb = (v & 1) + ((v >> 1) & 1) * 4 + (v >> 2) * 8 + hi * 2;
    f[v] = (int)rowdw[s0dw + kb];
  }
  return f;
}

__global__ __launch_bounds__(256) void k_pool(const float* __restrict__ x,
                                              _Float16* __restrict__ xp, int n) {
  int i = blockIdx.x * 256 + threadIdx.x;
  if (i >= n) return;
  const float4* p = (const float4*)(x + (size_t)i * 8);
  float4 a = p[0], c = p[1];
  float s = a.x + a.y + a.z + a.w + c.x + c.y + c.z + c.w;
  xp[i] = (_Float16)(s * 0.125f);
}

__global__ void k_bn(const float* __restrict__ g, const float* __restrict__ be,
                     const float* __restrict__ mn, const float* __restrict__ vr,
                     float* __restrict__ bns, float* __restrict__ bnb) {
  int i = threadIdx.x;
  float inv = g[i] * rsqrtf(vr[i] + 1e-5f);
  bns[i] = inv;
  bnb[i] = be[i] - mn[i] * inv;
}

__global__ __launch_bounds__(256) void k_qkv(const float* __restrict__ wqkv,
                                             const _Float16* __restrict__ xp,
                                             signed char* __restrict__ qkv) {
  __shared__ _Float16 sA[128 * 32];
  __shared__ _Float16 sBt[128 * 32];
  int blk = blockIdx.x;
  int b = blk / 48;
  int r = blk % 48;
  int mo = (r >> 2) * 128;
  int no = (r & 3) * 128;
  int tid = threadIdx.x, lane = tid & 31, w = tid >> 5, hi = lane >> 4;
  int wm = (w >> 2) * 64, wn = (w & 3) * 32;

  v8f zf = {0.f, 0.f, 0.f, 0.f, 0.f, 0.f, 0.f, 0.f};
  v8f acc[4][2] = {{zf, zf}, {zf, zf}, {zf, zf}, {zf, zf}};

  const size_t xpb = (size_t)b * CC * TP;
  for (int k0 = 0; k0 < CC; k0 += 32) {
    if (k0 + 32 < CC) {
      __builtin_prefetch(wqkv + (size_t)(mo + (tid >> 1)) * CC + (k0 + 32) + (tid & 1) * 16, 0, 1);
      __builtin_prefetch(xp + xpb + (size_t)(k0 + 32 + (tid >> 3)) * TP + no + (tid & 7) * 16, 0, 1);
    }
    __syncthreads();
    {
      int rr = tid >> 1, cb = (tid & 1) * 16;
      const float* src = wqkv + (size_t)(mo + rr) * CC + k0 + cb;
#pragma unroll
      for (int i = 0; i < 16; ++i) sA[rr * 32 + cb + i] = (_Float16)src[i];
    }
    {
      int rr = tid >> 3, cb = (tid & 7) * 16;
      const _Float16* src = xp + xpb + (size_t)(k0 + rr) * TP + no + cb;
#pragma unroll
      for (int i = 0; i < 16; ++i) sBt[(cb + i) * 32 + rr] = src[i];
    }
    __syncthreads();
#pragma unroll
    for (int nt = 0; nt < 2; ++nt) {
      v16h bf = b_frag16(sBt + (wn + nt * 16 + (lane & 15)) * 32, 0, hi);
#pragma unroll
      for (int mt = 0; mt < 4; ++mt) {
        v16h af = a_frag16(sA + (wm + mt * 16 + (lane & 15)) * 32, 0, hi);
        acc[mt][nt] = __builtin_amdgcn_wmma_f32_16x16x32_f16(
            false, af, false, bf, (short)0, acc[mt][nt], false, false);
      }
    }
  }
  const size_t qb = (size_t)b * O3 * TP;
#pragma unroll
  for (int mt = 0; mt < 4; ++mt)
#pragma unroll
    for (int nt = 0; nt < 2; ++nt)
#pragma unroll
      for (int j = 0; j < 8; ++j) {
        int o = mo + wm + mt * 16 + j + 8 * hi;
        int t = no + wn + nt * 16 + (lane & 15);
        qkv[qb + (size_t)o * TP + t] = (acc[mt][nt][j] >= 2.0f) ? 1 : 0;
      }
}

__global__ __launch_bounds__(256) void k_attn(const signed char* __restrict__ qkv,
                                              _Float16* __restrict__ attn_out) {
  __shared__ __align__(16) unsigned char smem[65536];
  int bh = blockIdx.x;
  int b = bh >> 3, h = bh & 7;
  int tid = threadIdx.x, lane = tid & 31, w = tid >> 5, hi = lane >> 4;
  const size_t base = (size_t)b * O3 * TP;
  const signed char* qg = qkv + base + (size_t)(h * DD) * TP;
  const signed char* kg = qkv + base + (size_t)(CC + h * DD) * TP;
  const signed char* vg = qkv + base + (size_t)(2 * CC + h * DD) * TP;

#if defined(HAVE_TDM)
  if (w == 0) {
    unsigned ldsbase = (unsigned)(size_t)(void*)smem;
    unsigned long long gak = (unsigned long long)(const void*)kg;
    unsigned long long gav = (unsigned long long)(const void*)vg;
    i32x8 g1 = {(int)(3u << 16),
                (int)(4096u << 16),
                (int)(1u << 16),
                (int)(4096u << 16),
                1,
                4096, 0, 0};
    i32x4 z4 = {0, 0, 0, 0};
    u32x4 g0k = {1u, ldsbase, (unsigned)gak,
                 (unsigned)(((gak >> 32) & 0x1ffffffu) | (2u << 30))};
    u32x4 g0v = {1u, ldsbase + 32768u, (unsigned)gav,
                 (unsigned)(((gav >> 32) & 0x1ffffffu) | (2u << 30))};
#  if defined(TDM_6ARG)
    i32x8 z8 = {0, 0, 0, 0, 0, 0, 0, 0};
    __builtin_amdgcn_tensor_load_to_lds(g0k, g1, z4, z4, z8, 0);
    __builtin_amdgcn_tensor_load_to_lds(g0v, g1, z4, z4, z8, 0);
#  else
    __builtin_amdgcn_tensor_load_to_lds(g0k, g1, z4, z4, 0);
    __builtin_amdgcn_tensor_load_to_lds(g0v, g1, z4, z4, 0);
#  endif
    __builtin_amdgcn_s_wait_tensorcnt(0);
  }
#else
  {
    const uint4* k4 = (const uint4*)kg;
    const uint4* v4 = (const uint4*)vg;
    uint4* s4 = (uint4*)smem;
    for (int i = tid; i < 2048; i += 256) { s4[i] = k4[i]; s4[2048 + i] = v4[i]; }
  }
#endif
  __builtin_prefetch(qg + (size_t)(tid >> 2) * TP + (tid & 3) * 128, 0, 1);
  __syncthreads();

  const unsigned* skd = (const unsigned*)smem;
  const unsigned* svd = (const unsigned*)(smem + 32768);
  v8i zi = {0, 0, 0, 0, 0, 0, 0, 0};
  v8i acc[2] = {zi, zi};
  int mts[2], nts[2];
#pragma unroll
  for (int u = 0; u < 2; ++u) {
    int T = w * 2 + u;
    mts[u] = (T >> 2) * 16;
    nts[u] = (T & 3) * 16;
  }
  for (int s0 = 0; s0 < 128; s0 += 16) {
#pragma unroll
    for (int u = 0; u < 2; ++u) {
      v8i af = a_frag8(skd + (mts[u] + (lane & 15)) * 128, s0, hi);
      v8i bf = b_frag8(svd + (nts[u] + (lane & 15)) * 128, s0, hi);
      acc[u] = __builtin_amdgcn_wmma_i32_16x16x64_iu8(false, af, false, bf,
                                                      acc[u], false, false);
    }
  }
  __syncthreads();

  _Float16* sKVT = (_Float16*)(smem + 32768);
  unsigned char* qT = (unsigned char*)smem;
#pragma unroll
  for (int u = 0; u < 2; ++u)
#pragma unroll
    for (int j = 0; j < 8; ++j) {
      int dp = mts[u] + j + 8 * hi;
      int d  = nts[u] + (lane & 15);
      sKVT[d * DD + dp] = (_Float16)(float)acc[u][j];
    }
  {
    int dp = tid >> 2;
    int tb = (tid & 3) * 128;
    const uint4* s4 = (const uint4*)(qg + dp * TP + tb);
#pragma unroll
    for (int vq = 0; vq < 8; ++vq) {
      uint4 dta = s4[vq];
      unsigned arr[4] = {dta.x, dta.y, dta.z, dta.w};
#pragma unroll
      for (int q2 = 0; q2 < 4; ++q2)
#pragma unroll
        for (int bb = 0; bb < 4; ++bb) {
          int i = vq * 16 + q2 * 4 + bb;
          qT[(tb + i) * DD + dp] = (unsigned char)(arr[q2] >> (8 * bb));
        }
    }
  }
  __syncthreads();

  const size_t ob = ((size_t)b * CC + h * DD) * TP;
#pragma unroll
  for (int nt = 0; nt < 4; ++nt) {
    int n0 = (w * 4 + nt) * 16;
    v8f zf = {0.f, 0.f, 0.f, 0.f, 0.f, 0.f, 0.f, 0.f};
    v8f oc[4] = {zf, zf, zf, zf};
#pragma unroll
    for (int k0 = 0; k0 < DD; k0 += 32) {
      v16h bf = b_frag16_i8(qT + (n0 + (lane & 15)) * DD, k0, hi);
#pragma unroll
      for (int mt = 0; mt < 4; ++mt) {
        v16h af = a_frag16(sKVT + (mt * 16 + (lane & 15)) * DD, k0, hi);
        oc[mt] = __builtin_amdgcn_wmma_f32_16x16x32_f16(
            false, af, false, bf, (short)0, oc[mt], false, false);
      }
    }
#pragma unroll
    for (int mt = 0; mt < 4; ++mt)
#pragma unroll
      for (int j = 0; j < 8; ++j) {
        int d = mt * 16 + j + 8 * hi;
        int t = n0 + (lane & 15);
        attn_out[ob + (size_t)d * TP + t] = (_Float16)(oc[mt][j] * 0.125f);
      }
  }
}

__global__ __launch_bounds__(256) void k_proj(const float* __restrict__ wout,
                                              const _Float16* __restrict__ attn,
                                              const float* __restrict__ bns,
                                              const float* __restrict__ bnb,
                                              float* __restrict__ out) {
  __shared__ _Float16 sA[128 * 32];
  __shared__ _Float16 sBt[128 * 32];
  int blk = blockIdx.x;
  int b = blk >> 4;
  int r = blk & 15;
  int mo = (r >> 2) * 128;
  int no = (r & 3) * 128;
  int tid = threadIdx.x, lane = tid & 31, w = tid >> 5, hi = lane >> 4;
  int wm = (w >> 2) * 64, wn = (w & 3) * 32;

  v8f zf = {0.f, 0.f, 0.f, 0.f, 0.f, 0.f, 0.f, 0.f};
  v8f acc[4][2] = {{zf, zf}, {zf, zf}, {zf, zf}, {zf, zf}};

  const size_t ab = (size_t)b * CC * TP;
  for (int k0 = 0; k0 < CC; k0 += 32) {
    if (k0 + 32 < CC) {
      __builtin_prefetch(wout + (size_t)(mo + (tid >> 1)) * CC + (k0 + 32) + (tid & 1) * 16, 0, 1);
      __builtin_prefetch(attn + ab + (size_t)(k0 + 32 + (tid >> 3)) * TP + no + (tid & 7) * 16, 0, 1);
    }
    __syncthreads();
    {
      int rr = tid >> 1, cb = (tid & 1) * 16;
      const float* src = wout + (size_t)(mo + rr) * CC + k0 + cb;
#pragma unroll
      for (int i = 0; i < 16; ++i) sA[rr * 32 + cb + i] = (_Float16)src[i];
    }
    {
      int rr = tid >> 3, cb = (tid & 7) * 16;
      const _Float16* src = attn + ab + (size_t)(k0 + rr) * TP + no + cb;
#pragma unroll
      for (int i = 0; i < 16; ++i) sBt[(cb + i) * 32 + rr] = src[i];
    }
    __syncthreads();
#pragma unroll
    for (int nt = 0; nt < 2; ++nt) {
      v16h bf = b_frag16(sBt + (wn + nt * 16 + (lane & 15)) * 32, 0, hi);
#pragma unroll
      for (int mt = 0; mt < 4; ++mt) {
        v16h af = a_frag16(sA + (wm + mt * 16 + (lane & 15)) * 32, 0, hi);
        acc[mt][nt] = __builtin_amdgcn_wmma_f32_16x16x32_f16(
            false, af, false, bf, (short)0, acc[mt][nt], false, false);
      }
    }
  }
#pragma unroll
  for (int mt = 0; mt < 4; ++mt)
#pragma unroll
    for (int nt = 0; nt < 2; ++nt)
#pragma unroll
      for (int j = 0; j < 8; ++j) {
        int o = mo + wm + mt * 16 + j + 8 * hi;
        int t = no + wn + nt * 16 + (lane & 15);
        float y = acc[mt][nt][j] * bns[o] + bnb[o];
        float s = (y >= 2.0f) ? 1.0f : 0.0f;
        float4 rv = {s, s, s, s};
        float* dst = out + ((size_t)b * CC + o) * T0 + (size_t)t * 8;
        *(float4*)dst = rv;
        *(float4*)(dst + 4) = rv;
      }
}

extern "C" void kernel_launch(void* const* d_in, const int* in_sizes, int n_in,
                              void* d_out, int out_size, void* d_ws, size_t ws_size,
                              hipStream_t stream) {
  const float* x    = (const float*)d_in[0];
  const float* wqkv = (const float*)d_in[1];
  const float* wout = (const float*)d_in[2];
  const float* g    = (const float*)d_in[3];
  const float* be   = (const float*)d_in[4];
  const float* mn   = (const float*)d_in[5];
  const float* vr   = (const float*)d_in[6];

  char* ws = (char*)d_ws;
  const size_t xp_bytes   = (size_t)BB * CC * TP * 2;
  const size_t qkv_bytes  = (size_t)BB * O3 * TP;
  const size_t attn_bytes = (size_t)BB * CC * TP * 2;
  _Float16*    xp   = (_Float16*)ws;
  signed char* qkv  = (signed char*)(ws + xp_bytes);
  _Float16*    attn = (_Float16*)(ws + xp_bytes + qkv_bytes);
  float*       bns  = (float*)(ws + xp_bytes + qkv_bytes + attn_bytes);
  float*       bnb  = bns + CC;

  int npool = BB * CC * TP;
  k_pool<<<dim3((npool + 255) / 256), dim3(256), 0, stream>>>(x, xp, npool);
  k_bn<<<dim3(1), dim3(CC), 0, stream>>>(g, be, mn, vr, bns, bnb);
  k_qkv<<<dim3(BB * 12 * 4), dim3(256), 0, stream>>>(wqkv, xp, qkv);
  k_attn<<<dim3(BB * HH), dim3(256), 0, stream>>>(qkv, attn);
  k_proj<<<dim3(BB * 4 * 4), dim3(256), 0, stream>>>(wout, attn, bns, bnb, (float*)d_out);
}
